// Part_Graph_28724741276433
// MI455X (gfx1250) — matrix operand order repaired and checked by reading, not compiled, not run on hardware
//
#include <hip/hip_runtime.h>

// ---------------------------------------------------------------------------
// CDNA5 (gfx1250): contractions via v_wmma_f32_16x16x32_bf16 with pre-packed
// bf16 operands (pixel-major activations, [9][Mpad][Kpad] weights).
// A panels staged block-wide into LDS via global_load_async_to_lds_b128
// (ASYNCcnt) and served by ds_load_b128.
// ---------------------------------------------------------------------------

typedef __attribute__((ext_vector_type(16))) __bf16 v16bf;
typedef __attribute__((ext_vector_type(8)))  __bf16 v8bf;
typedef __attribute__((ext_vector_type(8)))  float  v8f;

#define HW    9216
#define WDIM  96
#define NPART 6
#define NEDGE 10

__device__ __forceinline__ float sigm(float x) { return 1.f / (1.f + __expf(-x)); }
__device__ __forceinline__ int   imin(int a, int b) { return a < b ? a : b; }
__device__ __forceinline__ int   imax(int a, int b) { return a > b ? a : b; }

__device__ __forceinline__ v16bf zero16() { v16bf z = {}; return z; }

// Async-stage `total` bf16 elements from global `src` into LDS at `dst`
// (block-cooperative, 16B per lane issue). Caller must s_wait + barrier.
__device__ __forceinline__ void async_stage(const __bf16* __restrict__ src,
                                            __bf16* dst, int total) {
  for (int idx = threadIdx.x * 8; idx < total; idx += 256 * 8) {
    unsigned int laddr = (unsigned int)(unsigned long long)(void*)(dst + idx);
    unsigned long long gaddr = (unsigned long long)(const void*)(src + idx);
    asm volatile("global_load_async_to_lds_b128 %0, %1, off"
                 :: "v"(laddr), "v"(gaddr) : "memory");
  }
}
__device__ __forceinline__ void async_wait_barrier() {
  asm volatile("s_wait_asynccnt 0" ::: "memory");
  __syncthreads();
}

// B operand: pixel-major bf16. Lane covers K range [kb, kb+15] at pixel pidx.
__device__ __forceinline__ v16bf load_bvec(const __bf16* __restrict__ b0, int s0, int lim0,
                                           const __bf16* __restrict__ b1, int s1,
                                           int pidx, int kb) {
  if (kb < lim0) return *(const v16bf*)(b0 + (size_t)pidx * s0 + kb);
  int k2 = kb - lim0;
  if (b1 != nullptr && k2 < s1) return *(const v16bf*)(b1 + (size_t)pidx * s1 + k2);
  return zero16();
}

// A operand from packed [Mpad][Kpad] bf16 row (works on LDS or global ptrs):
// halves[0..7]=K[kBase+16*hi+0..7], halves[8..15]=K[kBase+16*hi+8..15]
__device__ __forceinline__ v16bf load_apk(const __bf16* row, int kBase, int hi) {
  v8bf lo = *(const v8bf*)(row + kBase + 16 * hi);
  v8bf hh = *(const v8bf*)(row + kBase + 16 * hi + 8);
  v16bf a;
#pragma unroll
  for (int j = 0; j < 8; ++j) { a[j] = lo[j]; a[j + 8] = hh[j]; }
  return a;
}

// ---------------------------------------------------------------------------
// Packing kernels
// ---------------------------------------------------------------------------
__global__ void packT_kernel(const float* __restrict__ src0, int n0,
                             const float* __restrict__ src1, int n1,
                             int stride, __bf16* __restrict__ out) {
  int i = blockIdx.x * blockDim.x + threadIdx.x;
  if (i >= HW * stride) return;
  int p = i / stride, c = i % stride;
  float v = 0.f;
  if (c < n0) v = src0[(size_t)c * HW + p];
  else if (src1 != nullptr && (c - n0) < n1) v = src1[(size_t)(c - n0) * HW + p];
  out[i] = (__bf16)v;
}

// conv weights f32 [Cout][Cin][3][3] -> bf16 [9][Mpad][Kpad] (zero padded)
__global__ void pack_wconv_kernel(const float* __restrict__ W, int Cout, int Cin,
                                  int Mpad, int Kpad, __bf16* __restrict__ out) {
  int i = blockIdx.x * blockDim.x + threadIdx.x;
  if (i >= 9 * Mpad * Kpad) return;
  int c = i % Kpad, m = (i / Kpad) % Mpad, k = i / (Kpad * Mpad);
  float v = (m < Cout && c < Cin) ? W[((size_t)m * Cin + c) * 9 + k] : 0.f;
  out[i] = (__bf16)v;
}

// 1x1 weights f32 [M][K] -> bf16 [Mpad][Kpad] (zero padded)
__global__ void pack_w1_kernel(const float* __restrict__ W, int M, int K,
                               int Mpad, int Kpad, __bf16* __restrict__ out) {
  int i = blockIdx.x * blockDim.x + threadIdx.x;
  if (i >= Mpad * Kpad) return;
  int c = i % Kpad, m = i / Kpad;
  float v = (m < M && c < K) ? W[(size_t)m * K + c] : 0.f;
  out[i] = (__bf16)v;
}

// ---------------------------------------------------------------------------
// 3x3 SAME conv (M=27 offset/mask conv) as 9 shifted 1x1 WMMA GEMMs.
// Per-tap A panel staged into LDS; all waves participate in barriers.
// ---------------------------------------------------------------------------
__global__ __launch_bounds__(256) void offconv_kernel(
    const __bf16* __restrict__ BT0, int s0, int lim0,
    const __bf16* __restrict__ BT1, int s1, int K,
    const __bf16* __restrict__ Wpk, int Kpad,   // [9][32][Kpad]
    const float* __restrict__ bias,
    float* __restrict__ Oout) {                 // [27][HW]
  __shared__ __bf16 shA[32 * 288];              // 18KB max panel
  const int M = 27;
  int jobs = 2 * WDIM * 6;
  int wid = (blockIdx.x * blockDim.x + threadIdx.x) >> 5;
  bool active = wid < jobs;
  int wz = active ? wid : 0;
  int mt  = wz / (WDIM * 6);
  int rem = wz % (WDIM * 6);
  int y   = rem / 6, x0 = (rem % 6) * 16;
  int lane = threadIdx.x & 31;
  int hi = lane >> 4, r = lane & 15, col = lane & 15;
  int mBase = mt * 16;
  int kchunks = (K + 31) / 32;
  int totalA = 32 * Kpad;

  v8f acc = {};
  for (int k = 0; k < 9; ++k) {
    async_stage(Wpk + (size_t)k * 32 * Kpad, shA, totalA);
    async_wait_barrier();
    if (active) {
      int ky = k / 3, kx = k % 3;
      int yy = y + ky - 1;
      int xx = x0 + col + kx - 1;
      bool pv = (yy >= 0 && yy < WDIM && xx >= 0 && xx < WDIM);
      int pidx = pv ? (yy * WDIM + xx) : 0;
      const __bf16* Arow = shA + (size_t)(mBase + r) * Kpad;   // LDS reads
      for (int kc = 0; kc < kchunks; ++kc) {
        int kBase = kc * 32;
        if ((kBase + 32) < lim0)
          __builtin_prefetch(BT0 + (size_t)pidx * s0 + kBase + 32, 0, 1);
        v16bf a = load_apk(Arow, kBase, hi);
        v16bf bl = load_bvec(BT0, s0, lim0, BT1, s1, pidx, kBase + 16 * hi);
        v16bf b = pv ? bl : zero16();
        acc = __builtin_amdgcn_wmma_f32_16x16x32_bf16(false, a, false, b,
                                                      (short)0, acc, false, false);
      }
    }
    __syncthreads();   // protect shA before next tap's staging
  }
  if (active) {
    int p = y * WDIM + x0 + col;
#pragma unroll
    for (int i = 0; i < 8; ++i) {
      int m = mBase + hi * 8 + i;
      if (m < M) Oout[(size_t)m * HW + p] = acc[i] + bias[m];
    }
  }
}

// ---------------------------------------------------------------------------
// Deformable conv main GEMM: bilinear-gathered bf16 B, fused affine+ReLU.
// Per-tap A panel staged into LDS.
// ---------------------------------------------------------------------------
__global__ __launch_bounds__(256) void dfconv_kernel(
    const __bf16* __restrict__ BT0, int s0, int lim0,
    const __bf16* __restrict__ BT1, int s1, int K, int Cout,
    const __bf16* __restrict__ Wpk, int Mpad, int Kpad,  // [9][Mpad][Kpad]
    const float* __restrict__ Off,                       // [27][HW]
    const float* __restrict__ scale, const float* __restrict__ shift,
    float* __restrict__ HoutF, __bf16* __restrict__ HoutT, int strideT) {
  __shared__ __bf16 shA[32 * 288];              // 18KB max panel
  int mtiles = (Cout + 15) / 16;
  int jobs = mtiles * WDIM * 6;
  int wid = (blockIdx.x * blockDim.x + threadIdx.x) >> 5;
  bool active = wid < jobs;
  int wz = active ? wid : 0;
  int mt  = wz / (WDIM * 6);
  int rem = wz % (WDIM * 6);
  int y   = rem / 6, x0 = (rem % 6) * 16;
  int lane = threadIdx.x & 31;
  int hi = lane >> 4, r = lane & 15, col = lane & 15;
  int mBase = mt * 16;
  int kchunks = (K + 31) / 32;
  int p = y * WDIM + x0 + col;
  float fy = (float)y, fx = (float)(x0 + col);
  int totalA = Mpad * Kpad;

  v8f acc = {};
  for (int k = 0; k < 9; ++k) {
    async_stage(Wpk + (size_t)k * Mpad * Kpad, shA, totalA);
    async_wait_barrier();
    if (active) {
      float off0 = Off[(size_t)(2 * k) * HW + p];
      float off1 = Off[(size_t)(2 * k + 1) * HW + p];
      float mk   = sigm(Off[(size_t)(18 + k) * HW + p]);
      float py = fy + (float)(k / 3 - 1) + off0;
      float px = fx + (float)(k % 3 - 1) + off1;
      float y0f = floorf(py), x0f = floorf(px);
      float wy = py - y0f, wx = px - x0f;
      int iy0 = (int)y0f, ix0 = (int)x0f;
      int iy1 = iy0 + 1, ix1 = ix0 + 1;
      float v00 = ((iy0 >= 0 && iy0 < WDIM && ix0 >= 0 && ix0 < WDIM) ? 1.f : 0.f) * (1.f - wy) * (1.f - wx) * mk;
      float v01 = ((iy0 >= 0 && iy0 < WDIM && ix1 >= 0 && ix1 < WDIM) ? 1.f : 0.f) * (1.f - wy) * wx * mk;
      float v10 = ((iy1 >= 0 && iy1 < WDIM && ix0 >= 0 && ix0 < WDIM) ? 1.f : 0.f) * wy * (1.f - wx) * mk;
      float v11 = ((iy1 >= 0 && iy1 < WDIM && ix1 >= 0 && ix1 < WDIM) ? 1.f : 0.f) * wy * wx * mk;
      int cy0 = imin(imax(iy0, 0), WDIM - 1), cx0 = imin(imax(ix0, 0), WDIM - 1);
      int cy1 = imin(imax(iy1, 0), WDIM - 1), cx1 = imin(imax(ix1, 0), WDIM - 1);
      int i00 = cy0 * WDIM + cx0, i01 = cy0 * WDIM + cx1;
      int i10 = cy1 * WDIM + cx0, i11 = cy1 * WDIM + cx1;
      const __bf16* Arow = shA + (size_t)(mBase + r) * Kpad;   // LDS reads
      for (int kc = 0; kc < kchunks; ++kc) {
        int kBase = kc * 32;
        v16bf a = load_apk(Arow, kBase, hi);
        int kb = kBase + 16 * hi;
        v16bf g00 = load_bvec(BT0, s0, lim0, BT1, s1, i00, kb);
        v16bf g01 = load_bvec(BT0, s0, lim0, BT1, s1, i01, kb);
        v16bf g10 = load_bvec(BT0, s0, lim0, BT1, s1, i10, kb);
        v16bf g11 = load_bvec(BT0, s0, lim0, BT1, s1, i11, kb);
        v16bf b;
#pragma unroll
        for (int t = 0; t < 16; ++t) {
          float s = v00 * (float)g00[t] + v01 * (float)g01[t]
                  + v10 * (float)g10[t] + v11 * (float)g11[t];
          b[t] = (__bf16)s;
        }
        acc = __builtin_amdgcn_wmma_f32_16x16x32_bf16(false, a, false, b,
                                                      (short)0, acc, false, false);
      }
    }
    __syncthreads();
  }
  if (active) {
#pragma unroll
    for (int i = 0; i < 8; ++i) {
      int m = mBase + hi * 8 + i;
      if (m < Cout) {
        float h = fmaxf(acc[i] * scale[m] + shift[m], 0.f);
        if (HoutF) HoutF[(size_t)m * HW + p] = h;
        if (HoutT) HoutT[(size_t)p * strideT + m] = (__bf16)h;
      }
    }
  }
}

// ---------------------------------------------------------------------------
// Edge attention + message + running max into xpp[dest]
// ---------------------------------------------------------------------------
__global__ void attmsg_kernel(const float* __restrict__ xpA, const float* __restrict__ xpB,
                              const float* __restrict__ aw, const float* __restrict__ ab,
                              const float* __restrict__ bw, const float* __restrict__ bb,
                              const float* __restrict__ H2, float* __restrict__ xppb) {
  int p = blockIdx.x * blockDim.x + threadIdx.x;
  if (p >= HW) return;
  float sa = ab[0], sb = bb[0];
#pragma unroll
  for (int c = 0; c < 10; ++c) {
    sa += aw[c] * xpA[(size_t)c * HW + p];
    sb += bw[c] * xpB[(size_t)c * HW + p];
  }
  float f = (1.f - sigm(sa)) * sigm(sb);
#pragma unroll
  for (int c = 0; c < 10; ++c) {
    float m = f * H2[(size_t)c * HW + p];
    float* dst = xppb + (size_t)c * HW + p;
    *dst = fmaxf(*dst, m);
  }
}

__global__ void fill_kernel(float* __restrict__ ptr, float val, int n) {
  int i = blockIdx.x * blockDim.x + threadIdx.x;
  if (i < n) ptr[i] = val;
}

// ---------------------------------------------------------------------------
// 1x1-conv WMMA GEMM, A panel async-staged to LDS.
// Out[m,p] = post(pscale[p] * sum_c A[m,c]*B[c,p])
// ---------------------------------------------------------------------------
__global__ __launch_bounds__(256) void gemm1x1_kernel(
    const __bf16* __restrict__ Wpk, int M, int K, int Kpad,  // [Mpad][Kpad]
    const __bf16* __restrict__ BT, int sB,                   // pixel-major bf16
    const float* __restrict__ pscale,
    const float* __restrict__ oscale, const float* __restrict__ oshift, int doRelu,
    const float* __restrict__ addTerm, float* __restrict__ Out) {
  __shared__ __bf16 shA[4096];   // max panel 16x256 (8KB)
  int mtiles = (M + 15) / 16;
  async_stage(Wpk, shA, mtiles * 16 * Kpad);
  async_wait_barrier();

  int wid = (blockIdx.x * blockDim.x + threadIdx.x) >> 5;
  if (wid >= mtiles * WDIM * 6) return;
  int mt  = wid / (WDIM * 6);
  int rem = wid % (WDIM * 6);
  int y   = rem / 6, x0 = (rem % 6) * 16;
  int lane = threadIdx.x & 31;
  int hi = lane >> 4, r = lane & 15, col = lane & 15;
  int mBase = mt * 16;
  int p = y * WDIM + x0 + col;
  int kchunks = (K + 31) / 32;
  const __bf16* Arow = shA + (size_t)(mBase + r) * Kpad;   // LDS -> ds_load

  v8f acc = {};
  for (int kc = 0; kc < kchunks; ++kc) {
    int kBase = kc * 32;
    if ((kBase + 32) < sB)
      __builtin_prefetch(BT + (size_t)p * sB + kBase + 32, 0, 1);
    v16bf a = load_apk(Arow, kBase, hi);
    v16bf b = load_bvec(BT, sB, sB, nullptr, 0, p, kBase + 16 * hi);
    acc = __builtin_amdgcn_wmma_f32_16x16x32_bf16(false, a, false, b,
                                                  (short)0, acc, false, false);
  }
  float ps = pscale ? pscale[p] : 1.f;
#pragma unroll
  for (int i = 0; i < 8; ++i) {
    int m = mBase + hi * 8 + i;
    if (m < M) {
      float v = acc[i] * ps;
      v = v * (oscale ? oscale[m] : 1.f) + oshift[m];
      if (doRelu) v = fmaxf(v, 0.f);
      if (addTerm) v += addTerm[(size_t)m * HW + p];
      Out[(size_t)m * HW + p] = v;
    }
  }
}

__global__ void att_kernel(const float* __restrict__ A_, const float* __restrict__ w2,
                           const float* __restrict__ b2, float* __restrict__ att) {
  int p = blockIdx.x * blockDim.x + threadIdx.x;
  if (p >= HW) return;
  float s = b2[0];
#pragma unroll
  for (int c = 0; c < 20; ++c) s += w2[c] * A_[(size_t)c * HW + p];
  att[p] = sigm(s);
}

// GRU cell with h0 == 0:  h = (1-sig(gz+bz)) * tanh(gn + sig(gr+br)*bn)
__global__ void gru_elt_kernel(const float* __restrict__ GI, const float* __restrict__ bhh,
                               float* __restrict__ outF, __bf16* __restrict__ outT) {
  int i = blockIdx.x * blockDim.x + threadIdx.x;
  if (i >= 10 * HW) return;
  int c = i / HW, p = i % HW;
  float rr = sigm(GI[(size_t)c * HW + p] + bhh[c]);
  float zz = sigm(GI[(size_t)(10 + c) * HW + p] + bhh[10 + c]);
  float nn = tanhf(GI[(size_t)(20 + c) * HW + p] + rr * bhh[20 + c]);
  float h = (1.f - zz) * nn;
  if (outF) outF[(size_t)c * HW + p] = h;
  if (outT) outT[(size_t)p * 16 + c] = (__bf16)h;
}

// ---------------------------------------------------------------------------
extern "C" void kernel_launch(void* const* d_in, const int* in_sizes, int n_in,
                              void* d_out, int out_size, void* d_ws, size_t ws_size,
                              hipStream_t stream) {
  (void)in_sizes; (void)n_in; (void)out_size; (void)ws_size;
  const float* pfea     = (const float*)d_in[0];
  const float* xp       = (const float*)d_in[1];
  const float* xh       = (const float*)d_in[2];
  const float* dp_ow1   = (const float*)d_in[3];
  const float* dp_ob1   = (const float*)d_in[4];
  const float* dp_w1    = (const float*)d_in[5];
  const float* dp_s1    = (const float*)d_in[6];
  const float* dp_b1    = (const float*)d_in[7];
  const float* dp_ow2   = (const float*)d_in[8];
  const float* dp_ob2   = (const float*)d_in[9];
  const float* dp_w2    = (const float*)d_in[10];
  const float* dp_s2    = (const float*)d_in[11];
  const float* dp_b2    = (const float*)d_in[12];
  const float* dp_aw    = (const float*)d_in[13];
  const float* dp_ab    = (const float*)d_in[14];
  const float* dp_bw    = (const float*)d_in[15];
  const float* dp_bb    = (const float*)d_in[16];
  const float* dc_w1    = (const float*)d_in[17];
  const float* dc_s1    = (const float*)d_in[18];
  const float* dc_b1    = (const float*)d_in[19];
  const float* dc_w2    = (const float*)d_in[20];
  const float* dc_b2    = (const float*)d_in[21];
  const float* dc_wfh   = (const float*)d_in[22];
  const float* dc_sfh   = (const float*)d_in[23];
  const float* dc_bfh   = (const float*)d_in[24];
  const float* gru_wih1 = (const float*)d_in[25];
  const float* gru_bih1 = (const float*)d_in[26];
  const float* gru_bhh1 = (const float*)d_in[28];
  const float* gru_wih2 = (const float*)d_in[29];
  const float* gru_bih2 = (const float*)d_in[30];
  const float* gru_bhh2 = (const float*)d_in[32];

  float* ws  = (float*)d_ws;
  // fp32 region
  float* O1  = ws;                    // 27*HW
  float* O2  = ws + 248832;           // 27*HW
  float* H2  = ws + 497664;           // 10*HW
  float* XPP = ws + 589824;           // 60*HW
  float* A_  = ws + 1142784;          // 20*HW
  float* ATT = ws + 1327104;          // HW
  float* GI  = ws + 1336320;          // 30*HW
  // bf16 region (all offsets multiples of 16 elements -> 32B aligned)
  __bf16* bfb  = (__bf16*)(ws + 1612800);
  __bf16* PT   = bfb + 0;             // [HW][256]
  __bf16* XT   = bfb + 2359296;       // 6 x [HW][16]
  __bf16* XHT  = bfb + 3244032;       // 2 x [HW][16]
  __bf16* H1T  = bfb + 3538944;       // [HW][32]
  __bf16* CATT = bfb + 3833856;       // [HW][32]
  __bf16* XINT = bfb + 4128768;       // [HW][32]
  __bf16* H1GT = bfb + 4423680;       // [HW][16]
  __bf16* WA1  = bfb + 4571136;       // [9][32][288]
  __bf16* WD1  = bfb + 4654080;       // [9][32][288]
  __bf16* WA2  = bfb + 4737024;       // [9][32][32]
  __bf16* WD2  = bfb + 4746240;       // [9][16][32]
  __bf16* W1X  = bfb + 4750848;       // [32][32]
  __bf16* WFH  = bfb + 4751872;       // [16][256]
  __bf16* WI1  = bfb + 4755968;       // [32][32]
  __bf16* WI2  = bfb + 4756992;       // [32][32]

  float* outXp  = (float*)d_out;
  float* outMsg = outXp + (size_t)NPART * 10 * HW;

  auto blks = [](int n) { return (n + 255) / 256; };
  auto wblocks = [](int jobs) { return (jobs * 32 + 255) / 256; };
  static const int EA[NEDGE] = {0, 1, 1, 1, 2, 2, 3, 4, 4, 5};
  static const int EB[NEDGE] = {1, 0, 2, 4, 1, 3, 2, 1, 5, 4};

  // ---- one-time packs ----
  packT_kernel<<<blks(HW * 256), 256, 0, stream>>>(pfea, 256, nullptr, 0, 256, PT);
  for (int j = 0; j < NPART; ++j)
    packT_kernel<<<blks(HW * 16), 256, 0, stream>>>(xp + (size_t)j * 10 * HW, 10,
                                                    nullptr, 0, 16, XT + (size_t)j * HW * 16);
  for (int h = 0; h < 2; ++h)
    packT_kernel<<<blks(HW * 16), 256, 0, stream>>>(xh + (size_t)h * 10 * HW, 10,
                                                    nullptr, 0, 16, XHT + (size_t)h * HW * 16);
  fill_kernel<<<blks(NPART * 10 * HW), 256, 0, stream>>>(XPP, -3.0e38f, NPART * 10 * HW);

  // ---- per-edge deformable message pipeline ----
  for (int e = 0; e < NEDGE; ++e) {
    const __bf16* XTa = XT + (size_t)EA[e] * HW * 16;
    const float*  xpa = xp + (size_t)EA[e] * 10 * HW;
    const float*  xpb = xp + (size_t)EB[e] * 10 * HW;
    pack_wconv_kernel<<<blks(9 * 32 * 288), 256, 0, stream>>>(
        dp_ow1 + (size_t)e * 27 * 266 * 9, 27, 266, 32, 288, WA1);
    pack_wconv_kernel<<<blks(9 * 32 * 288), 256, 0, stream>>>(
        dp_w1 + (size_t)e * 20 * 266 * 9, 20, 266, 32, 288, WD1);
    pack_wconv_kernel<<<blks(9 * 32 * 32), 256, 0, stream>>>(
        dp_ow2 + (size_t)e * 27 * 20 * 9, 27, 20, 32, 32, WA2);
    pack_wconv_kernel<<<blks(9 * 16 * 32), 256, 0, stream>>>(
        dp_w2 + (size_t)e * 10 * 20 * 9, 10, 20, 16, 32, WD2);

    offconv_kernel<<<wblocks(2 * 576), 256, 0, stream>>>(
        PT, 256, 256, XTa, 16, 266, WA1, 288, dp_ob1 + e * 27, O1);
    dfconv_kernel<<<wblocks(2 * 576), 256, 0, stream>>>(
        PT, 256, 256, XTa, 16, 266, 20, WD1, 32, 288, O1,
        dp_s1 + e * 20, dp_b1 + e * 20, nullptr, H1T, 32);
    offconv_kernel<<<wblocks(2 * 576), 256, 0, stream>>>(
        H1T, 32, 32, nullptr, 0, 20, WA2, 32, dp_ob2 + e * 27, O2);
    dfconv_kernel<<<wblocks(1 * 576), 256, 0, stream>>>(
        H1T, 32, 32, nullptr, 0, 20, 10, WD2, 16, 32, O2,
        dp_s2 + e * 10, dp_b2 + e * 10, H2, nullptr, 0);
    attmsg_kernel<<<blks(HW), 256, 0, stream>>>(
        xpa, xpb, dp_aw + e * 10, dp_ab + e, dp_bw + e * 10, dp_bb + e,
        H2, XPP + (size_t)EB[e] * 10 * HW);
  }

  // ---- per-node context attention -> msgs ----
  for (int j = 0; j < NPART; ++j) {
    const float*  hf  = xh + (size_t)((j < 4) ? 0 : 1) * 10 * HW;
    const float*  xpj = xp + (size_t)j * 10 * HW;
    pack_w1_kernel<<<blks(32 * 32), 256, 0, stream>>>(dc_w1 + j * 400, 20, 20, 32, 32, W1X);
    packT_kernel<<<blks(HW * 32), 256, 0, stream>>>(hf, 10, xpj, 10, 32, CATT);
    gemm1x1_kernel<<<wblocks(2 * 576), 256, 0, stream>>>(
        W1X, 20, 20, 32, CATT, 32, nullptr, dc_s1 + j * 20, dc_b1 + j * 20, 1, nullptr, A_);
    att_kernel<<<blks(HW), 256, 0, stream>>>(A_, dc_w2 + j * 20, dc_b2 + j, ATT);
    pack_w1_kernel<<<blks(16 * 256), 256, 0, stream>>>(dc_wfh + j * 2560, 10, 256, 16, 256, WFH);
    gemm1x1_kernel<<<wblocks(1 * 576), 256, 0, stream>>>(
        WFH, 10, 256, 256, PT, 256, ATT, dc_sfh + j * 10, dc_bfh + j * 10, 1,
        XPP + (size_t)j * 10 * HW, outMsg + (size_t)j * 10 * HW);
  }

  // ---- per-node 2-step GRU ----
  for (int j = 0; j < NPART; ++j) {
    const float* xpj  = xp + (size_t)j * 10 * HW;
    const float* msgj = outMsg + (size_t)j * 10 * HW;
    pack_w1_kernel<<<blks(32 * 32), 256, 0, stream>>>(gru_wih1 + j * 600, 30, 20, 32, 32, WI1);
    packT_kernel<<<blks(HW * 32), 256, 0, stream>>>(xpj, 10, msgj, 10, 32, XINT);
    gemm1x1_kernel<<<wblocks(2 * 576), 256, 0, stream>>>(
        WI1, 30, 20, 32, XINT, 32, nullptr, nullptr, gru_bih1 + j * 30, 0, nullptr, GI);
    gru_elt_kernel<<<blks(10 * HW), 256, 0, stream>>>(GI, gru_bhh1 + j * 30, nullptr, H1GT);
    pack_w1_kernel<<<blks(32 * 32), 256, 0, stream>>>(gru_wih2 + j * 300, 30, 10, 32, 32, WI2);
    gemm1x1_kernel<<<wblocks(2 * 576), 256, 0, stream>>>(
        WI2, 30, 10, 32, H1GT, 16, nullptr, nullptr, gru_bih2 + j * 30, 0, nullptr, GI);
    gru_elt_kernel<<<blks(10 * HW), 256, 0, stream>>>(GI, gru_bhh2 + j * 30,
                                                      outXp + (size_t)j * 10 * HW, nullptr);
  }
}
